// EICLayer2_11484742550200
// MI455X (gfx1250) — compile-verified, matrix operand order checked by reading, not run on hardware
//
#include <hip/hip_runtime.h>
#include <hip/hip_bf16.h>

typedef __bf16 bf16_t;
typedef __attribute__((ext_vector_type(16))) __bf16 v16bf;
typedef __attribute__((ext_vector_type(8)))  __bf16 v8bf;
typedef __attribute__((ext_vector_type(8)))  float  v8f;

typedef __attribute__((ext_vector_type(4))) unsigned tdm_v4u;
typedef __attribute__((ext_vector_type(8))) int      tdm_v8i;
typedef __attribute__((ext_vector_type(4))) int      tdm_v4i;

#define BATCH           16384
#define NCORE           16
#define ROWS_PER_BLOCK  128          // 8 waves * 16 rows
#define MBLOCKS         (BATCH / ROWS_PER_BLOCK)   // 128

// TDM staging: ROCm 7.2 (clang-22) = 5-arg builtin; amdgpu-toolchain
// (clang-23, therock-10.0) = 6-arg builtin (extra zero-filled v8i).
#if __has_builtin(__builtin_amdgcn_tensor_load_to_lds)
#define EIC_HAVE_TDM 1
#else
#define EIC_HAVE_TDM 0
#endif

// ---------------------------------------------------------------------------
// Workspace layout (bf16 elements), weights pre-split into hi/lo bf16 planes
// stored in exact WMMA B-fragment order.
//   core cid<12 : K=256 -> planeElems = 8*8192 = 65536, 2 planes
//   core cid>=12: K=192 -> planeElems = 6*8192 = 49152, 2 planes
// Fragment (ks,t) occupies 512 elems (1 KiB): elem i of lane L at
//   (ks*16+t)*512 + (i>>3)*256 + L*8 + (i&7)
// 16B chunks arranged [chunk][lane] -> wave32 ds_load_b128 is only 2-way
// bank conflicted (the minimum for 16B/lane on 64 banks).
// ---------------------------------------------------------------------------
__device__ inline size_t core_ws_base(int cid) {
  return (cid < 12) ? (size_t)cid * 131072u
                    : (size_t)12 * 131072u + (size_t)(cid - 12) * 98304u;
}

// Split fp32 weights into (hi, lo) bf16 planes in B-fragment layout.
// B-fragment element mapping for V_WMMA_F32_16X16X32_BF16 (wave32):
//   B is 32(K) x 16(N); element (k,n):
//     lane = ((k>>3)&1)*16 + n
//     vgpr = (k>>4)*4 + ((k&7)>>1), half = k&1, i = vgpr*2 + half
__global__ __launch_bounds__(256)
void split_weights_kernel(const float* __restrict__ W256,
                          const float* __restrict__ W192,
                          bf16_t* __restrict__ wf) {
  const int cid = blockIdx.x;
  const int K = (cid < 12) ? 256 : 192;
  const int planeElems = (K >> 5) * 8192;
  const float* __restrict__ W = (cid < 12)
      ? (W256 + (size_t)cid * 256 * 256)
      : (W192 + (size_t)(cid - 12) * 256 * 192);
  bf16_t* __restrict__ dst = wf + core_ws_base(cid);
  const int total = 256 * K;             // g-major, f contiguous (torch [out,in])
  for (int e = threadIdx.x; e < total; e += blockDim.x) {
    const int g = e / K;
    const int f = e - g * K;
    const float w = W[e];
    const bf16_t h = (bf16_t)w;
    const float rem = w - (float)h;
    const bf16_t l = (bf16_t)rem;
    const int ks = f >> 5, k = f & 31, t = g >> 4, n = g & 15;
    const int lane = (((k >> 3) & 1) << 4) | n;
    const int v    = ((k >> 4) << 2) | ((k & 7) >> 1);
    const int i    = (v << 1) | (k & 1);
    const int idx  = ((ks * 16 + t) << 9) + ((i >> 3) << 8) + (lane << 3) + (i & 7);
    dst[idx]              = h;
    dst[planeElems + idx] = l;
  }
}

// ---------------------------------------------------------------------------
// Fused GEMM + (sigmoid(y - 0.5*rowmax)) epilogue.
// Block = 8 waves = one core (r,c) x 128 batch rows; wave = 16 rows x 256 cols.
// ---------------------------------------------------------------------------
__global__ __launch_bounds__(256)
void eic_gemm_kernel(const float* __restrict__ x,
                     const bf16_t* __restrict__ wf,
                     float* __restrict__ out) {
  extern __shared__ bf16_t smem[];           // [2 * planeElems] hi plane, lo plane

  const int cid  = blockIdx.x >> 7;          // / MBLOCKS(128)
  const int mblk = blockIdx.x & 127;
  const int r = cid >> 2, c = cid & 3;
  const int nks = (r < 3) ? 8 : 6;           // K/32
  const int planeElems = nks * 8192;

#if EIC_HAVE_TDM
  // One TDM DMA per block: linear 2*planeElems*2 bytes global -> LDS.
  if ((threadIdx.x >> 5) == 0) {             // wave 0 only (uniform branch)
    const unsigned long long ga =
        (unsigned long long)(const void*)(wf + core_ws_base(cid));
    const unsigned words = (unsigned)((planeElems * 2 * 2) >> 3); // 8B units
    // D# group 0: count=1, lds_addr=0 (dynamic LDS base), global_addr, type=2
    tdm_v4u g0;
    g0[0] = 1u;                              // count = 1 valid descriptor
    g0[1] = 0u;                              // lds_addr (bytes)
    g0[2] = (unsigned)(ga & 0xFFFFFFFFull);  // global_addr[31:0]
    g0[3] = (unsigned)((ga >> 32) & 0x01FFFFFFull) | (2u << 30); // [56:32]|type=2
    // D# group 1: data_size=3 (8B), 1-D tile: tensor_dim0=tile_dim0=words
    tdm_v8i g1;
    g1[0] = (int)(3u << 16);                 // workgroup_mask=0, data_size=8B
    g1[1] = (int)((words & 0xFFFFu) << 16);  // tensor_dim0[15:0] @ bit48
    g1[2] = (int)(((words >> 16) & 0xFFFFu) | (1u << 16)); // td0[31:16], td1=1
    g1[3] = (int)((words & 0xFFFFu) << 16);  // td1[31:16]=0, tile_dim0 @ bit112
    g1[4] = 0;                               // tile_dim1=0 (unused), tile_dim2=0
    g1[5] = (int)words;                      // tensor_dim0_stride[31:0]
    g1[6] = 0;                               // stride0[47:32], stride1[15:0]
    g1[7] = 0;                               // tensor_dim1_stride[47:16]
    tdm_v4i z4 = {0, 0, 0, 0};               // groups 2/3 unused (<=2-D tensor)
#if __clang_major__ >= 23
    tdm_v8i z8 = {0, 0, 0, 0, 0, 0, 0, 0};
    __builtin_amdgcn_tensor_load_to_lds(g0, g1, z4, z4, z8, 0);
#else
    __builtin_amdgcn_tensor_load_to_lds(g0, g1, z4, z4, 0);
#endif
    __builtin_amdgcn_s_wait_tensorcnt(0);
  }
#else
  // Fallback: cooperative copy global -> LDS.
  {
    const uint4* __restrict__ src = (const uint4*)(wf + core_ws_base(cid));
    uint4* dst = (uint4*)smem;
    const int nchunks = (planeElems * 2) >> 3;   // 16B chunks
    for (int i = threadIdx.x; i < nchunks; i += 256) dst[i] = src[i];
  }
#endif
  __syncthreads();

  const int lane  = threadIdx.x & 31;
  const int wv    = threadIdx.x >> 5;
  const int laneN = lane & 15;
  const int kmid  = lane >> 4;               // which half of the 16-K block
  const int m0    = mblk * ROWS_PER_BLOCK + wv * 16;
  const int mA    = m0 + laneN;              // batch row this lane feeds into A

  v8f acc[16];
  #pragma unroll
  for (int t = 0; t < 16; ++t) acc[t] = v8f{0.f, 0.f, 0.f, 0.f, 0.f, 0.f, 0.f, 0.f};

  for (int ks = 0; ks < nks; ++ks) {
    // Chunk permutation: feature f = l1c*64 + p  maps to x[b, r, l1c, c*64 + p].
    // A 32-wide K window (32-aligned) stays inside one l1c block of 64.
    const int l1c = ks >> 1;
    const float* __restrict__ rp =
        x + ((size_t)(mA * 4 + r) * 4 + l1c) * 256 + c * 64 + ((ks & 1) << 5);
    // A fragment: elem i of lane <-> k = (i/8)*16 + kmid*8 + (i%8)
    const float4 fa0 = *(const float4*)(rp + kmid * 8);
    const float4 fa1 = *(const float4*)(rp + kmid * 8 + 4);
    const float4 fb0 = *(const float4*)(rp + 16 + kmid * 8);
    const float4 fb1 = *(const float4*)(rp + 16 + kmid * 8 + 4);
    const float fv[16] = {fa0.x, fa0.y, fa0.z, fa0.w, fa1.x, fa1.y, fa1.z, fa1.w,
                          fb0.x, fb0.y, fb0.z, fb0.w, fb1.x, fb1.y, fb1.z, fb1.w};
    v16bf ahi, alo;
    #pragma unroll
    for (int i = 0; i < 16; ++i) {
      const bf16_t h = (bf16_t)fv[i];
      const float rem = fv[i] - (float)h;
      ahi[i] = h;
      alo[i] = (bf16_t)rem;
    }

    #pragma unroll
    for (int t = 0; t < 16; ++t) {
      const int fb = ((ks * 16 + t) << 9) + (lane << 3);
      const v8bf h0 = *(const v8bf*)(smem + fb);
      const v8bf h1 = *(const v8bf*)(smem + fb + 256);
      const v8bf l0 = *(const v8bf*)(smem + planeElems + fb);
      const v8bf l1 = *(const v8bf*)(smem + planeElems + fb + 256);
      const v16bf bh = __builtin_shufflevector(h0, h1, 0, 1, 2, 3, 4, 5, 6, 7,
                                               8, 9, 10, 11, 12, 13, 14, 15);
      const v16bf bl = __builtin_shufflevector(l0, l1, 0, 1, 2, 3, 4, 5, 6, 7,
                                               8, 9, 10, 11, 12, 13, 14, 15);
      // bf16x3: hi*hi + hi*lo + lo*hi, accumulated in f32 (drops ~2^-18 term)
      acc[t] = __builtin_amdgcn_wmma_f32_16x16x32_bf16(false, ahi, false, bh,
                                                       (short)0, acc[t], false, false);
      acc[t] = __builtin_amdgcn_wmma_f32_16x16x32_bf16(false, ahi, false, bl,
                                                       (short)0, acc[t], false, false);
      acc[t] = __builtin_amdgcn_wmma_f32_16x16x32_bf16(false, alo, false, bh,
                                                       (short)0, acc[t], false, false);
    }
  }

  // C/D fragment: vgpr j, lanes0-15 -> row j, lanes16-31 -> row j+8, col=laneN.
  // Row max over all 256 outputs: reduce over n-tiles in-register, then over
  // the 16 lanes of each half (xor 8,4,2,1 stays within a half in wave32).
  float pm[8];
  #pragma unroll
  for (int j = 0; j < 8; ++j) {
    float m = acc[0][j];
    #pragma unroll
    for (int t = 1; t < 16; ++t) m = fmaxf(m, acc[t][j]);
    m = fmaxf(m, __shfl_xor(m, 8, 32));
    m = fmaxf(m, __shfl_xor(m, 4, 32));
    m = fmaxf(m, __shfl_xor(m, 2, 32));
    m = fmaxf(m, __shfl_xor(m, 1, 32));
    pm[j] = m;
  }

  #pragma unroll
  for (int j = 0; j < 8; ++j) {
    const int row = m0 + j + (kmid << 3);
    float* __restrict__ orow =
        out + ((size_t)(row * 4 + r) * 4 + c) * 256 + laneN;
    #pragma unroll
    for (int t = 0; t < 16; ++t) {
      const float z = acc[t][j] - 0.5f * pm[j];
      orow[t * 16] = 1.0f / (1.0f + __expf(-z));
    }
  }
}

// ---------------------------------------------------------------------------
extern "C" void kernel_launch(void* const* d_in, const int* in_sizes, int n_in,
                              void* d_out, int out_size, void* d_ws, size_t ws_size,
                              hipStream_t stream) {
  (void)in_sizes; (void)n_in; (void)out_size; (void)ws_size;
  const float* x    = (const float*)d_in[0];
  const float* W256 = (const float*)d_in[1];
  const float* W192 = (const float*)d_in[2];
  float*   out = (float*)d_out;
  bf16_t*  wf  = (bf16_t*)d_ws;   // 1,966,080 bf16 = 3.93 MB of scratch

  // 1) split weights to bf16 hi/lo, fragment layout (cheap: 4 MB of writes)
  split_weights_kernel<<<dim3(NCORE), dim3(256), 0, stream>>>(W256, W192, wf);

  // 2) fused GEMM + epilogue; 256 KiB dynamic LDS per block (<= 320 KiB/WGP)
  const size_t shmem = 2u * 65536u * sizeof(bf16_t);  // 256 KiB
  hipFuncSetAttribute((const void*)eic_gemm_kernel,
                      hipFuncAttributeMaxDynamicSharedMemorySize, (int)shmem);
  eic_gemm_kernel<<<dim3(NCORE * MBLOCKS), dim3(256), shmem, stream>>>(x, wf, out);
}
